// SelfAttention_5033701670933
// MI455X (gfx1250) — matrix operand order repaired and checked by reading, not compiled, not run on hardware
//
#include <hip/hip_runtime.h>

// ---------------------------------------------------------------------------
// Self-attention for MI455X (gfx1250, wave32, WMMA 16x16x32 f16)
//   q = x@Wq.T ; k = x@Wk.T ; v = x@Wv.T      (f16 WMMA, f32 accum)
//   out = softmax(q@k.T / 8) @ v              (flash-style, online softmax)
// ---------------------------------------------------------------------------

typedef __attribute__((ext_vector_type(16))) _Float16 v16h;
typedef __attribute__((ext_vector_type(8)))  _Float16 v8h;
typedef __attribute__((ext_vector_type(8)))  float    v8f;

#define SEQ 8192
#define DIN 512
#define DH  64

// log2(e) / sqrt(DH) : fold softmax scale into base-2 exponentials
#define SCALE_LOG2 (1.4426950408889634f * 0.125f)

// workspace layout in _Float16 elements
#define XH_OFF  ((size_t)0)                      // x  f16  [SEQ, DIN]   8 MB
#define WQ_OFF  (XH_OFF + (size_t)SEQ * DIN)     // Wq f16  [DH, DIN]
#define WK_OFF  (WQ_OFF + (size_t)DH * DIN)
#define WV_OFF  (WK_OFF + (size_t)DH * DIN)
#define QH_OFF  (WV_OFF + (size_t)DH * DIN)      // Q  f16  [SEQ, DH]
#define KH_OFF  (QH_OFF + (size_t)SEQ * DH)      // K  f16  [SEQ, DH]
#define VT_OFF  (KH_OFF + (size_t)SEQ * DH)      // V^T f16 [DH, SEQ]

static __device__ __forceinline__ v16h cat8(v8h lo, v8h hi) {
  return __builtin_shufflevector(lo, hi, 0,1,2,3,4,5,6,7,8,9,10,11,12,13,14,15);
}

// ---------------------------------------------------------------------------
// f32 -> f16 conversion (grid-stride)
// ---------------------------------------------------------------------------
__global__ void cvt_f32_to_f16(const float* __restrict__ src,
                               _Float16* __restrict__ dst, int n) {
  int i = blockIdx.x * blockDim.x + threadIdx.x;
  int stride = gridDim.x * blockDim.x;
  for (; i < n; i += stride) dst[i] = (_Float16)src[i];
}

// ---------------------------------------------------------------------------
// One wave computes a 16x16 tile of (x @ W^T).
//   xh : [SEQ, DIN] f16 row-major   (A operand)
//   wh : [DH,  DIN] f16 row-major   (B operand: B[k][n] = W[n][k] -> W rows
//                                    are contiguous in the WMMA K dimension)
//   transpose_out = 0 : out[SEQ, DH] row-major (Q, K)
//   transpose_out = 1 : out[DH, SEQ]           (V^T, so P@V B-frags are
//                                               contiguous 16B loads later)
// ---------------------------------------------------------------------------
__global__ __launch_bounds__(32)
void qkv_proj(const _Float16* __restrict__ xh,
              const _Float16* __restrict__ wh,
              _Float16* __restrict__ outh,
              int transpose_out) {
  const int lane = threadIdx.x & 31;
  const int l16  = lane & 15;
  const int h    = lane >> 4;
  const int m0   = blockIdx.x * 16;   // rows of x
  const int n0   = blockIdx.y * 16;   // output columns (rows of W)

  const _Float16* arow = xh + (size_t)(m0 + l16) * DIN;
  const _Float16* brow = wh + (size_t)(n0 + l16) * DIN;

  v8f c = {};
  #pragma unroll 4
  for (int kk = 0; kk < DIN; kk += 32) {
    // A fragment: row m0+l16, K elems kk+16h .. kk+16h+15 (contiguous)
    v16h a = cat8(*(const v8h*)(arow + kk + 16 * h),
                  *(const v8h*)(arow + kk + 16 * h + 8));
    // B fragment: col n0+l16, K elems kk+16h .. kk+16h+15 (contiguous)
    v16h b = cat8(*(const v8h*)(brow + kk + 16 * h),
                  *(const v8h*)(brow + kk + 16 * h + 8));
    c = __builtin_amdgcn_wmma_f32_16x16x32_f16(false, a, false, b,
                                               (short)0, c, false, false);
  }

  if (!transpose_out) {
    #pragma unroll
    for (int r = 0; r < 8; ++r)
      outh[(size_t)(m0 + r + 8 * h) * DH + n0 + l16] = (_Float16)c[r];
  } else {
    #pragma unroll
    for (int r = 0; r < 8; ++r)
      outh[(size_t)(n0 + l16) * SEQ + m0 + r + 8 * h] = (_Float16)c[r];
  }
}

// ---------------------------------------------------------------------------
// Flash attention: each wave owns 16 query rows, streams all 8192 keys in
// blocks of 32. Per key block: 4 WMMAs for S = Q@K^T (two 16-key tiles x two
// 32-wide head chunks), base-2 online softmax with shfl_xor butterflies in
// each 16-lane half, C-layout -> A-layout relayout of P through LDS, then
// 4 WMMAs for O += P@V.
// ---------------------------------------------------------------------------
__global__ __launch_bounds__(128)
void flash_attn(const _Float16* __restrict__ qh,
                const _Float16* __restrict__ kh,
                const _Float16* __restrict__ vth,
                float* __restrict__ out) {
  __shared__ __align__(16) _Float16 Plds[4][16][32];   // per-wave 1KB staging

  const int lane = threadIdx.x & 31;
  const int wid  = threadIdx.x >> 5;
  const int l16  = lane & 15;
  const int h    = lane >> 4;
  const int m0   = (blockIdx.x * 4 + wid) * 16;        // query strip

  // Q fragments for head chunks [0,32) and [32,64), loaded once
  const _Float16* qrow = qh + (size_t)(m0 + l16) * DH;
  const v16h qa0 = cat8(*(const v8h*)(qrow + 16 * h),
                        *(const v8h*)(qrow + 16 * h + 8));
  const v16h qa1 = cat8(*(const v8h*)(qrow + 32 + 16 * h),
                        *(const v8h*)(qrow + 32 + 16 * h + 8));

  v8f o0 = {}, o1 = {}, o2 = {}, o3 = {};
  float M[8], Lsum[8];
  #pragma unroll
  for (int r = 0; r < 8; ++r) { M[r] = -__builtin_inff(); Lsum[r] = 0.0f; }

  for (int j = 0; j < SEQ; j += 32) {
    if (j + 32 < SEQ) {  // pull next key block toward the caches
      __builtin_prefetch(kh + (size_t)(j + 32 + l16) * DH, 0, 0);
    }

    // ---- S = Q @ K^T for keys [j, j+32) --------------------------------
    const _Float16* krow0 = kh + (size_t)(j + l16) * DH;        // keys j..j+15
    const _Float16* krow1 = kh + (size_t)(j + 16 + l16) * DH;   // keys j+16..j+31
    v8f s1 = {}, s2 = {};
    {
      v16h b00 = cat8(*(const v8h*)(krow0 + 16 * h),
                      *(const v8h*)(krow0 + 16 * h + 8));        // head 0..31
      v16h b01 = cat8(*(const v8h*)(krow0 + 32 + 16 * h),
                      *(const v8h*)(krow0 + 32 + 16 * h + 8));   // head 32..63
      s1 = __builtin_amdgcn_wmma_f32_16x16x32_f16(false, qa0, false, b00,
                                                  (short)0, s1, false, false);
      s1 = __builtin_amdgcn_wmma_f32_16x16x32_f16(false, qa1, false, b01,
                                                  (short)0, s1, false, false);
      v16h b10 = cat8(*(const v8h*)(krow1 + 16 * h),
                      *(const v8h*)(krow1 + 16 * h + 8));
      v16h b11 = cat8(*(const v8h*)(krow1 + 32 + 16 * h),
                      *(const v8h*)(krow1 + 32 + 16 * h + 8));
      s2 = __builtin_amdgcn_wmma_f32_16x16x32_f16(false, qa0, false, b10,
                                                  (short)0, s2, false, false);
      s2 = __builtin_amdgcn_wmma_f32_16x16x32_f16(false, qa1, false, b11,
                                                  (short)0, s2, false, false);
    }

    // ---- online softmax (base-2) --------------------------------------
    // C layout: element r of lane -> row (r + 8h), col (l16) / (16+l16)
    #pragma unroll
    for (int r = 0; r < 8; ++r) {
      float t1 = s1[r] * SCALE_LOG2;
      float t2 = s2[r] * SCALE_LOG2;
      float mx = fmaxf(t1, t2);
      #pragma unroll
      for (int msk = 1; msk <= 8; msk <<= 1)
        mx = fmaxf(mx, __shfl_xor(mx, msk, 32));     // reduce 16-lane half
      float Mn = fmaxf(M[r], mx);
      float p1 = exp2f(t1 - Mn);
      float p2 = exp2f(t2 - Mn);
      float corr = exp2f(M[r] - Mn);                 // 0 on first iteration
      float ps = p1 + p2;
      #pragma unroll
      for (int msk = 1; msk <= 8; msk <<= 1)
        ps += __shfl_xor(ps, msk, 32);
      Lsum[r] = Lsum[r] * corr + ps;
      M[r] = Mn;
      o0[r] *= corr; o1[r] *= corr; o2[r] *= corr; o3[r] *= corr;
      Plds[wid][r + 8 * h][l16]      = (_Float16)p1;
      Plds[wid][r + 8 * h][16 + l16] = (_Float16)p2;
    }

    // cross-lane hand-off through LDS: keep ds stores before ds loads
    asm volatile("" ::: "memory");

    // P as A-fragment: row l16, K elems 16h..16h+15 (contiguous)
    v16h pa = cat8(*(const v8h*)(&Plds[wid][l16][16 * h]),
                   *(const v8h*)(&Plds[wid][l16][16 * h + 8]));

    // ---- O += P @ V  (V^T rows are contiguous in the key dimension) ----
    #pragma unroll
    for (int t = 0; t < 4; ++t) {
      const _Float16* vcol = vth + (size_t)(16 * t + l16) * SEQ + j + 16 * h;
      v16h vb = cat8(*(const v8h*)(vcol), *(const v8h*)(vcol + 8));
      v8f& o = (t == 0) ? o0 : (t == 1) ? o1 : (t == 2) ? o2 : o3;
      o = __builtin_amdgcn_wmma_f32_16x16x32_f16(false, pa, false, vb,
                                                 (short)0, o, false, false);
    }
  }

  // ---- normalize and write out[SEQ, DH] f32 ----------------------------
  #pragma unroll
  for (int r = 0; r < 8; ++r) {
    float inv = 1.0f / Lsum[r];
    float* orow = out + (size_t)(m0 + r + 8 * h) * DH;
    orow[l16]      = o0[r] * inv;
    orow[16 + l16] = o1[r] * inv;
    orow[32 + l16] = o2[r] * inv;
    orow[48 + l16] = o3[r] * inv;
  }
}

// ---------------------------------------------------------------------------
extern "C" void kernel_launch(void* const* d_in, const int* in_sizes, int n_in,
                              void* d_out, int out_size, void* d_ws, size_t ws_size,
                              hipStream_t stream) {
  const float* x  = (const float*)d_in[0];
  const float* Wq = (const float*)d_in[1];
  const float* Wk = (const float*)d_in[2];
  const float* Wv = (const float*)d_in[3];
  float* out = (float*)d_out;

  _Float16* ws  = (_Float16*)d_ws;
  _Float16* xh  = ws + XH_OFF;
  _Float16* wqh = ws + WQ_OFF;
  _Float16* wkh = ws + WK_OFF;
  _Float16* wvh = ws + WV_OFF;
  _Float16* q_h = ws + QH_OFF;
  _Float16* k_h = ws + KH_OFF;
  _Float16* vt  = ws + VT_OFF;

  // 1) convert inputs to f16
  cvt_f32_to_f16<<<2048, 256, 0, stream>>>(x,  xh,  SEQ * DIN);
  cvt_f32_to_f16<<<128,  256, 0, stream>>>(Wq, wqh, DH * DIN);
  cvt_f32_to_f16<<<128,  256, 0, stream>>>(Wk, wkh, DH * DIN);
  cvt_f32_to_f16<<<128,  256, 0, stream>>>(Wv, wvh, DH * DIN);

  // 2) QKV projections (V stored transposed for the P@V B-operand)
  dim3 pg(SEQ / 16, DH / 16);
  qkv_proj<<<pg, 32, 0, stream>>>(xh, wqh, q_h, 0);
  qkv_proj<<<pg, 32, 0, stream>>>(xh, wkh, k_h, 0);
  qkv_proj<<<pg, 32, 0, stream>>>(xh, wvh, vt, 1);

  // 3) flash attention: 512 query strips of 16 rows, 4 waves/block
  flash_attn<<<SEQ / 64, 128, 0, stream>>>(q_h, k_h, vt, out);
}
